// LiteralE_KBLN_25065429139739
// MI455X (gfx1250) — matrix operand order repaired and checked by reading, not compiled, not run on hardware
//
#include <hip/hip_runtime.h>

typedef float  v8f   __attribute__((ext_vector_type(8)));
typedef __bf16 v16bf __attribute__((ext_vector_type(16)));
typedef __bf16 v8bf  __attribute__((ext_vector_type(8)));

#define E_N 40000
#define D_N 200
#define L_N 100
#define B_N 32
#define KC  300      // D+L
#define KP  320      // stage-1 K padded to 10 chunks of 32
#define NT1 13       // stage-1 N tiles (208 cols)
#define DP  224      // stage-2 K padded to 7 chunks of 32
#define TILE_E 64

// workspace layout (bytes)
#define WB_OFF  0
#define WB_SIZE (NT1*10*512*2)        // 133120: W^T bf16, B-frag packed
#define QB_OFF  (WB_OFF + WB_SIZE)
#define QB_SIZE (B_N*DP*2)            // 14336: Q = e1_emb*rel_emb, bf16 row-major padded
#define AB_OFF  (QB_OFF + QB_SIZE)
#define AB_SIZE (B_N*L_N*4)           // 12800: lit_b - c
#define WV_OFF  (AB_OFF + AB_SIZE)    // 12800: nf_w[rel]
#define S2_OFF  (WV_OFF + AB_SIZE)    // 400:  -log2e/var

__device__ __forceinline__ __bf16 f2bf(float f) {
    union { float f; unsigned u; } v; v.f = f;
    unsigned r = v.u + 0x7FFFu + ((v.u >> 16) & 1u);   // RNE
    union { unsigned short s; __bf16 b; } o;
    o.s = (unsigned short)(r >> 16);
    return o.b;
}

// ---------------------------------------------------------------- prep kernel
__global__ __launch_bounds__(256) void lite_kbln_prep(
    const int* __restrict__ e1, const int* __restrict__ rel,
    const float* __restrict__ emb_e, const float* __restrict__ emb_rel,
    const float* __restrict__ num_lit, const float* __restrict__ nlit_norm,
    const float* __restrict__ W, const float* __restrict__ b_lit,
    const float* __restrict__ c, const float* __restrict__ var,
    const float* __restrict__ nf_w,
    __bf16* __restrict__ Wb, __bf16* __restrict__ Qb,
    float* __restrict__ Ab, float* __restrict__ Wv, float* __restrict__ S2)
{
    const int tid = threadIdx.x;
    const int blk = blockIdx.x;
    if (blk < NT1) {
        // pack W^T (K=300->320, N=200->208) into per-(kc,nt) 32-lane x 16-val B fragments
        const int nt = blk;
        for (int idx = tid; idx < 10*512; idx += 256) {
            int kc = idx >> 9;
            int r  = idx & 511;
            int j  = r >> 4, i = r & 15;               // lane j, element i
            int n  = nt*16 + (j & 15);
            int k  = kc*32 + ((j >> 4) ? 16 : 0) + i;  // lanes 0-15: K 0..15, 16-31: K 16..31
            float v = (n < D_N && k < KC) ? W[n*KC + k] : 0.f;
            Wb[(kc*NT1 + nt)*512 + r] = f2bf(v);
        }
    } else if (blk == NT1) {
        // Q[b][d] = (cat(emb_e[e1],nlit_norm[e1]) @ W^T + b)[d] * emb_rel[rel][d]
        for (int o = tid; o < B_N*D_N; o += 256) {
            int b = o / D_N, d = o % D_N;
            int e = e1[b], rr = rel[b];
            float acc = b_lit[d];
            const float* wrow = W + d*KC;
            const float* er = emb_e + e*D_N;
            for (int k = 0; k < D_N; ++k) acc += er[k] * wrow[k];
            const float* lr = nlit_norm + e*L_N;
            for (int k = 0; k < L_N; ++k) acc += lr[k] * wrow[D_N + k];
            Qb[b*DP + d] = f2bf(acc * emb_rel[rr*D_N + d]);
        }
        for (int o = tid; o < B_N*(DP - D_N); o += 256) {   // zero K pad
            int b = o / (DP - D_N), d = D_N + o % (DP - D_N);
            Qb[b*DP + d] = f2bf(0.f);
        }
    } else if (blk == NT1 + 1) {
        for (int o = tid; o < B_N*L_N; o += 256) {
            int b = o / L_N, l = o % L_N;
            Ab[o] = num_lit[e1[b]*L_N + l] - c[l];
            Wv[o] = nf_w[rel[b]*L_N + l];
        }
    } else {
        if (tid < L_N) S2[tid] = -1.4426950408889634f / var[tid];
    }
}

// ---------------------------------------------------------------- fused main
__global__ __launch_bounds__(256) void lite_kbln_main(
    const float* __restrict__ emb_e, const float* __restrict__ nlit_norm,
    const float* __restrict__ num_lit, const float* __restrict__ b_lit,
    const __bf16* __restrict__ Wb, const __bf16* __restrict__ Qb,
    const float* __restrict__ Ab, const float* __restrict__ Wv,
    const float* __restrict__ S2, float* __restrict__ out)
{
    __shared__ union {
        __bf16 x16[16*KP];                 // stage-1 A staging (one 16-row slab)
        float  score[B_N*TILE_E];          // stage-2 output (reused after stage 1)
    } uXS;
    __shared__ __bf16 sZ[TILE_E*DP];       // e2 tile, bf16, rows padded to 224
    __shared__ float  sLit[TILE_E*L_N];    // num_lit rows for this entity tile

    const int tid    = threadIdx.x;
    const int wave   = tid >> 5;
    const int lane   = tid & 31;
    const int lr     = lane & 15;
    const int lh     = lane >> 4;
    const int e_base = blockIdx.x * TILE_E;

    // phase 0: stage literal rows (contiguous), zero sZ pad cols 208..223
    for (int idx = tid; idx < TILE_E*L_N; idx += 256)
        sLit[idx] = num_lit[e_base*L_N + idx];
    for (int idx = tid; idx < TILE_E*16; idx += 256) {
        int row = idx >> 4, col = 208 + (idx & 15);
        sZ[row*DP + col] = f2bf(0.f);
    }

    // ---- stage 1: Z(64x208) = X(64x320) @ W^T, 16 rows at a time -----------
    for (int mt = 0; mt < 4; ++mt) {
        __syncthreads();
        for (int idx = tid; idx < 16*KP; idx += 256) {
            int r = idx / KP, k = idx % KP;
            int e = e_base + mt*16 + r;
            float x = (k < D_N) ? emb_e[e*D_N + k]
                    : (k < KC ? nlit_norm[e*L_N + (k - D_N)] : 0.f);
            uXS.x16[idx] = f2bf(x);
        }
        __syncthreads();
        for (int nt = wave; nt < NT1; nt += 8) {
            v8f acc = {};
            for (int kc = 0; kc < 10; ++kc) {
                union { v16bf v; v8bf h[2]; } fa, fb;
                const __bf16* pa = &uXS.x16[lr*KP + kc*32 + (lh ? 16 : 0)];
                fa.h[0] = *(const v8bf*)pa;          // K k..k+7
                fa.h[1] = *(const v8bf*)(pa + 8);    // K k+8..k+15
                const __bf16* pb = Wb + (kc*NT1 + nt)*512 + lane*16;
                fb.h[0] = *(const v8bf*)pb;
                fb.h[1] = *(const v8bf*)(pb + 8);
                acc = __builtin_amdgcn_wmma_f32_16x16x32_bf16(
                        false, fa.v, false, fb.v, (short)0, acc, false, false);
            }
            int n = nt*16 + lr;
            float bias = (n < D_N) ? b_lit[n] : 0.f;
            int rowbase = mt*16 + (lh << 3);
            #pragma unroll
            for (int q = 0; q < 8; ++q)
                sZ[(rowbase + q)*DP + n] = f2bf(acc[q] + bias);
        }
    }
    __syncthreads();

    // ---- stage 2: score_l(32x64) = Q(32x224) @ Z^T -------------------------
    {
        const int mt2 = wave >> 2, nt2 = wave & 3;   // 2 b-tiles x 4 e-tiles = 8 waves
        v8f acc = {};
        for (int kc = 0; kc < 7; ++kc) {
            union { v16bf v; v8bf h[2]; } fa, fb;
            const __bf16* pa = Qb + (mt2*16 + lr)*DP + kc*32 + (lh ? 16 : 0);
            fa.h[0] = *(const v8bf*)pa;
            fa.h[1] = *(const v8bf*)(pa + 8);
            const __bf16* pb = &sZ[(nt2*16 + lr)*DP + kc*32 + (lh ? 16 : 0)];
            fb.h[0] = *(const v8bf*)pb;              // K contiguous along Z row
            fb.h[1] = *(const v8bf*)(pb + 8);
            acc = __builtin_amdgcn_wmma_f32_16x16x32_bf16(
                    false, fa.v, false, fb.v, (short)0, acc, false, false);
        }
        int e_loc  = nt2*16 + lr;
        int bbase  = mt2*16 + (lh << 3);
        #pragma unroll
        for (int q = 0; q < 8; ++q)
            uXS.score[(bbase + q)*TILE_E + e_loc] = acc[q];
    }
    __syncthreads();

    // ---- stage 3: RBF literal score + sigmoid ------------------------------
    {
        const int e_loc = tid & 63, bq = tid >> 6;   // bq uniform per wave
        float accn[8];
        #pragma unroll
        for (int i = 0; i < 8; ++i) accn[i] = 0.f;
        for (int l = 0; l < L_N; l += 4) {
            float4 xv = *(const float4*)&sLit[e_loc*L_N + l];
            float4 sv = *(const float4*)&S2[l];
            #pragma unroll
            for (int bi = 0; bi < 8; ++bi) {
                int b = bq*8 + bi;
                float4 av = *(const float4*)&Ab[b*L_N + l];
                float4 w4 = *(const float4*)&Wv[b*L_N + l];
                float t;
                t = av.x - xv.x; accn[bi] += w4.x * __builtin_amdgcn_exp2f(t*t*sv.x);
                t = av.y - xv.y; accn[bi] += w4.y * __builtin_amdgcn_exp2f(t*t*sv.y);
                t = av.z - xv.z; accn[bi] += w4.z * __builtin_amdgcn_exp2f(t*t*sv.z);
                t = av.w - xv.w; accn[bi] += w4.w * __builtin_amdgcn_exp2f(t*t*sv.w);
            }
        }
        #pragma unroll
        for (int bi = 0; bi < 8; ++bi) {
            int b = bq*8 + bi;
            float val = uXS.score[b*TILE_E + e_loc] + accn[bi];
            float ex  = __builtin_amdgcn_exp2f(-val * 1.4426950408889634f);
            out[b*E_N + e_base + e_loc] = __builtin_amdgcn_rcpf(1.f + ex);
        }
    }
}

// ---------------------------------------------------------------- launcher
extern "C" void kernel_launch(void* const* d_in, const int* in_sizes, int n_in,
                              void* d_out, int out_size, void* d_ws, size_t ws_size,
                              hipStream_t stream) {
    const int*   e1        = (const int*)  d_in[0];
    const int*   rel       = (const int*)  d_in[1];
    const float* emb_e     = (const float*)d_in[2];
    const float* emb_rel   = (const float*)d_in[3];
    const float* num_lit   = (const float*)d_in[4];
    const float* nlit_norm = (const float*)d_in[5];
    const float* W_lit     = (const float*)d_in[6];
    const float* b_lit     = (const float*)d_in[7];
    const float* c         = (const float*)d_in[8];
    const float* var       = (const float*)d_in[9];
    const float* nf_w      = (const float*)d_in[10];
    float* out = (float*)d_out;

    char* ws = (char*)d_ws;
    __bf16* Wb = (__bf16*)(ws + WB_OFF);
    __bf16* Qb = (__bf16*)(ws + QB_OFF);
    float*  Ab = (float*)(ws + AB_OFF);
    float*  Wv = (float*)(ws + WV_OFF);
    float*  S2 = (float*)(ws + S2_OFF);

    lite_kbln_prep<<<dim3(NT1 + 3), dim3(256), 0, stream>>>(
        e1, rel, emb_e, emb_rel, num_lit, nlit_norm, W_lit, b_lit, c, var, nf_w,
        Wb, Qb, Ab, Wv, S2);

    lite_kbln_main<<<dim3(E_N / TILE_E), dim3(256), 0, stream>>>(
        emb_e, nlit_norm, num_lit, b_lit, Wb, Qb, Ab, Wv, S2, out);
}
